// CausalMultiHeadSelfAttention_80539226734672
// MI455X (gfx1250) — matrix operand order repaired and checked by reading, not compiled, not run on hardware
//
#include <hip/hip_runtime.h>

// ---------------------------------------------------------------------------
// Causal MHA (B=2,S=2048,D=1024,H=16,dk=64) with QK-RMSNorm + tanh softcap.
// All GEMMs on v_wmma_f32_16x16x32_bf16 (f32 accumulate); softmax/softcap in
// branch-free f32 (v_exp_f32 / v_rcp_f32). V is produced pre-transposed per
// head so attention B-fragments are contiguous 16B loads. GEMM A-tiles are
// staged to LDS with gfx1250 async global->LDS copies when available.
// ---------------------------------------------------------------------------

typedef __attribute__((ext_vector_type(16))) __bf16 v16bf;
typedef __attribute__((ext_vector_type(8)))  __bf16 v8bf;
typedef __attribute__((ext_vector_type(8)))  float  v8f;
typedef __attribute__((ext_vector_type(4)))  int    v4i;

#define DEV static __device__ __forceinline__

constexpr int   Bc    = 2;
constexpr int   Sc    = 2048;
constexpr int   Dc    = 1024;
constexpr int   Hc    = 16;
constexpr int   DKc   = 64;
constexpr int   Mrows = Bc * Sc;                  // 4096
constexpr float SOFTCAP = 50.0f;
constexpr float EPSF    = 1.1920929e-07f;
constexpr float LOG2E   = 1.4426950408889634f;
// softcap folded constant: (1/sqrt(dk)) * 2*log2e / SOFTCAP
constexpr float CAPC    = 0.125f * 2.0f * LOG2E / SOFTCAP;

DEV v16bf cat16(v8bf a, v8bf b) {
  return __builtin_shufflevector(a, b, 0,1,2,3,4,5,6,7,8,9,10,11,12,13,14,15);
}

DEV v8f wmma_bf16(v16bf a, v16bf b, v8f c) {
  return __builtin_amdgcn_wmma_f32_16x16x32_bf16(
      /*neg_a=*/false, a, /*neg_b=*/false, b,
      /*c_mod=*/(short)0, c, /*reuse_a=*/false, /*reuse_b=*/false);
}

DEV float fast_exp2(float x) { return __builtin_amdgcn_exp2f(x); }
DEV float fast_rcp(float x)  { return __builtin_amdgcn_rcpf(x); }
DEV float fast_rsq(float x)  {
#if __has_builtin(__builtin_amdgcn_rsqf)
  return __builtin_amdgcn_rsqf(x);
#else
  return rsqrtf(x);
#endif
}

// branch-free softcap of a raw score s: SOFTCAP * tanh(s*scale/SOFTCAP)
DEV float softcap(float s) {
  return SOFTCAP - 2.0f * SOFTCAP * fast_rcp(1.0f + fast_exp2(s * CAPC));
}

// reduce across the 16 lanes of each wave32 half (rows live per-half)
DEV float redmax16(float v) {
  v = fmaxf(v, __shfl_xor(v, 1, 32));
  v = fmaxf(v, __shfl_xor(v, 2, 32));
  v = fmaxf(v, __shfl_xor(v, 4, 32));
  v = fmaxf(v, __shfl_xor(v, 8, 32));
  return v;
}
DEV float redsum16(float v) {
  v += __shfl_xor(v, 1, 32);
  v += __shfl_xor(v, 2, 32);
  v += __shfl_xor(v, 4, 32);
  v += __shfl_xor(v, 8, 32);
  return v;
}

// ---------------------------------------------------------------------------
// gfx1250 async global -> LDS copy (ASYNCcnt path), with portable fallback.
// Builtin signature (from toolchain diagnostic): param0 = int4 __device__* (AS1),
// param1 = LDS side, then imm offset, imm cpol.
// ---------------------------------------------------------------------------
#if __has_builtin(__builtin_amdgcn_global_load_async_to_lds_b128) && \
    __has_builtin(__builtin_amdgcn_s_wait_asynccnt)
#define HAVE_ASYNC_LDS 1
#else
#define HAVE_ASYNC_LDS 0
#endif

DEV void async_copy_b128(__bf16* lds_dst, const __bf16* gsrc) {
#if HAVE_ASYNC_LDS
  // integer round trips (inttoptr) are legal across address spaces;
  // LDS byte address = low 32 bits of the generic pointer per ISA aperture rule.
  __builtin_amdgcn_global_load_async_to_lds_b128(
      (v4i __attribute__((address_space(1)))*)(size_t)gsrc,
      (v4i __attribute__((address_space(3)))*)(size_t)(unsigned)(size_t)lds_dst,
      0, 0);
#else
  *(v8bf*)lds_dst = *(const v8bf*)gsrc;
#endif
}
DEV void async_join() {
#if HAVE_ASYNC_LDS
  __builtin_amdgcn_s_wait_asynccnt(0);
#endif
}

// ---------------------------------------------------------------------------
// elementwise f32 -> bf16
// ---------------------------------------------------------------------------
__global__ void __launch_bounds__(256)
cast_f32_bf16(const float* __restrict__ in, __bf16* __restrict__ out, int n) {
  int i = blockIdx.x * blockDim.x + threadIdx.x;
  int stride = gridDim.x * blockDim.x;
  for (; i < n; i += stride) out[i] = (__bf16)in[i];
}

// ---------------------------------------------------------------------------
// Y[m,n] = sum_k X[m,k] * W[n,k]   (X: MxK bf16 row-major, W: NxK bf16 row-major)
// Block = 128 threads (4 waves); the 16-row A tile (16xK, contiguous) is
// async-staged to LDS once; each wave computes a 16x128 strip (8 accumulators,
// A-frag reused 8x from LDS). grid = (M/16, N/512).
// MODE: 0 = f32 row-major, 1 = bf16 row-major, 2 = bf16 transposed-per-head
//       (Vt[b,h,dk,s], i.e. keys contiguous — the attention B-frag layout).
// ---------------------------------------------------------------------------
template <int MODE>
__global__ void __launch_bounds__(128)
gemm_xwT(const __bf16* __restrict__ X, const __bf16* __restrict__ W,
         void* __restrict__ Yv, int M, int N, int K) {
  __shared__ __bf16 Xs[16 * Dc];          // 32 KB A tile
  const int lane = threadIdx.x & 31;
  const int wave = threadIdx.x >> 5;
  const int lo = lane & 15;
  const int hi = lane >> 4;
  const int m0 = blockIdx.x * 16;
  const int n0 = blockIdx.y * 512 + wave * 128;

  // ---- cooperative async stage of the A tile (rows m0..m0+15 contiguous)
  {
    const __bf16* xg = X + (size_t)m0 * K;
    const int nchunk = (16 * K) / (8 * 128);   // 16 for K=1024
    for (int it = 0; it < nchunk; ++it) {
      const int e = (it * 128 + threadIdx.x) * 8;
      async_copy_b128(Xs + e, xg + e);
    }
    async_join();
    __syncthreads();
  }

  v8f acc[8];
#pragma unroll
  for (int t = 0; t < 8; ++t) acc[t] = {};

  // A-frag: lane row = lo, K = hi*16 .. +15 (contiguous, matches B-frag)
  const __bf16* xrow = Xs + lo * K + hi * 16;

  for (int k0 = 0; k0 < K; k0 += 32) {
    v8bf a0 = *(const v8bf*)(xrow + k0);          // ds_load_b128
    v8bf a1 = *(const v8bf*)(xrow + k0 + 8);
    v16bf a = cat16(a0, a1);
    __builtin_prefetch(W + (size_t)(n0 + lo) * K + k0 + 256, 0, 1);
#pragma unroll
    for (int t = 0; t < 8; ++t) {
      // B-frag: lane column = n0+t*16+lo, K = k0 + hi*16 .. +15 (contig row of W)
      const __bf16* wrow = W + (size_t)(n0 + t * 16 + lo) * K + k0 + hi * 16;
      v8bf b0 = *(const v8bf*)(wrow);
      v8bf b1 = *(const v8bf*)(wrow + 8);
      v16bf b = cat16(b0, b1);
      acc[t] = wmma_bf16(a, b, acc[t]);
    }
  }

#pragma unroll
  for (int t = 0; t < 8; ++t) {
    const int n = n0 + t * 16 + lo;
#pragma unroll
    for (int r = 0; r < 8; ++r) {
      const int m = m0 + r + hi * 8;  // C/D layout: lanes 0-15 row r, 16-31 row r+8
      if (MODE == 0) {
        ((float*)Yv)[(size_t)m * N + n] = acc[t][r];
      } else if (MODE == 1) {
        ((__bf16*)Yv)[(size_t)m * N + n] = (__bf16)acc[t][r];
      } else {
        // transposed-per-head: Vt[((b*H + h)*64 + dk)*S + s]
        const int b  = m >> 11;          // m / Sc
        const int s  = m & (Sc - 1);
        const int h  = n >> 6;           // n / 64
        const int dk = n & 63;
        ((__bf16*)Yv)[(((size_t)b * Hc + h) * DKc + dk) * Sc + s] = (__bf16)acc[t][r];
      }
    }
  }
}

// ---------------------------------------------------------------------------
// per-head RMSNorm over dk=64, in place on bf16 (stats in f32).
// one wave per head-vector; block = 256 threads = 8 vectors.
// ---------------------------------------------------------------------------
__global__ void __launch_bounds__(256)
rmsnorm_heads(__bf16* __restrict__ buf, int nvec) {
  const int lane = threadIdx.x & 31;
  const int wave = threadIdx.x >> 5;
  const int v = blockIdx.x * 8 + wave;
  if (v >= nvec) return;
  __bf16* p = buf + (size_t)v * 64 + lane * 2;
  const float a = (float)p[0];
  const float b = (float)p[1];
  float ss = a * a + b * b;
#pragma unroll
  for (int m = 16; m >= 1; m >>= 1) ss += __shfl_xor(ss, m, 32);
  const float r = fast_rsq(ss * (1.0f / 64.0f) + EPSF);
  p[0] = (__bf16)(a * r);
  p[1] = (__bf16)(b * r);
}

// ---------------------------------------------------------------------------
// Flash attention, one wave per (b,h, 16-row q tile), kv tiles of 32 keys.
// Q,K bf16 [B,S,D] (head h at cols h*64..); Vt bf16 [B,H,dk,S] (pre-transposed
// so V B-frags are contiguous). O bf16 [B,S,D].
// ---------------------------------------------------------------------------
__global__ void __launch_bounds__(32)
flash_attn(const __bf16* __restrict__ Q, const __bf16* __restrict__ Kh,
           const __bf16* __restrict__ Vt, __bf16* __restrict__ O) {
  __shared__ __bf16 Ps[16 * 32];   // P tile repack (C-layout -> A-layout)

  const int lane = threadIdx.x;
  const int lo = lane & 15;
  const int hi = lane >> 4;
  const int q0 = blockIdx.x * 16;
  const int bh = blockIdx.y;
  const int b = bh >> 4;
  const int h = bh & 15;
  const size_t base   = (size_t)b * Sc * Dc + (size_t)h * DKc;          // Q/K/O
  const size_t vtbase = ((size_t)b * Hc + h) * DKc * Sc;                // Vt

  // Q fragments: two K-halves of dk=64
  const __bf16* qrow = Q + base + (size_t)(q0 + lo) * Dc + hi * 16;
  v16bf qa[2];
#pragma unroll
  for (int c = 0; c < 2; ++c) {
    v8bf a0 = *(const v8bf*)(qrow + c * 32);
    v8bf a1 = *(const v8bf*)(qrow + c * 32 + 8);
    qa[c] = cat16(a0, a1);
  }

  float mr[8], lr[8];
  v8f acc[4];
#pragma unroll
  for (int r = 0; r < 8; ++r) { mr[r] = -1e30f; lr[r] = 0.0f; }
#pragma unroll
  for (int t = 0; t < 4; ++t) acc[t] = {};

  for (int j = 0; j <= q0 + 15; j += 32) {
    // ---- scores for keys [j, j+16) and [j+16, j+32)
    v8f s0 = {}, s1 = {};
#pragma unroll
    for (int c = 0; c < 2; ++c) {
      const __bf16* k0p = Kh + base + (size_t)(j + lo) * Dc + c * 32 + hi * 16;
      v16bf kb0 = cat16(*(const v8bf*)k0p, *(const v8bf*)(k0p + 8));
      s0 = wmma_bf16(qa[c], kb0, s0);
      const __bf16* k1p = Kh + base + (size_t)(j + 16 + lo) * Dc + c * 32 + hi * 16;
      v16bf kb1 = cat16(*(const v8bf*)k1p, *(const v8bf*)(k1p + 8));
      s1 = wmma_bf16(qa[c], kb1, s1);
    }

    // ---- branch-free softcap + causal mask + online softmax (f32)
    float p0[8], p1[8], alpha[8];
#pragma unroll
    for (int r = 0; r < 8; ++r) {
      const int row = q0 + r + hi * 8;
      float x0 = softcap(s0[r]);
      float x1 = softcap(s1[r]);
      x0 = (j + lo      > row) ? -1e30f : x0;
      x1 = (j + 16 + lo > row) ? -1e30f : x1;
      const float tmax = redmax16(fmaxf(x0, x1));
      const float mn = fmaxf(mr[r], tmax);
      alpha[r] = fast_exp2((mr[r] - mn) * LOG2E);
      mr[r] = mn;
      p0[r] = fast_exp2((x0 - mn) * LOG2E);
      p1[r] = fast_exp2((x1 - mn) * LOG2E);
      const float rs = redsum16(p0[r] + p1[r]);
      lr[r] = lr[r] * alpha[r] + rs;
    }
#pragma unroll
    for (int t = 0; t < 4; ++t)
#pragma unroll
      for (int r = 0; r < 8; ++r) acc[t][r] *= alpha[r];

    // ---- repack P (16x32) via LDS into A-fragment layout
    __syncthreads();
#pragma unroll
    for (int r = 0; r < 8; ++r) {
      const int row = r + hi * 8;
      Ps[row * 32 + lo]      = (__bf16)p0[r];
      Ps[row * 32 + 16 + lo] = (__bf16)p1[r];
    }
    __syncthreads();
    v8bf pa0 = *(const v8bf*)(Ps + lo * 32 + hi * 16);
    v8bf pa1 = *(const v8bf*)(Ps + lo * 32 + hi * 16 + 8);
    v16bf pa = cat16(pa0, pa1);

    // ---- O += P @ V : V B-frags are contiguous in Vt (keys along memory)
#pragma unroll
    for (int t = 0; t < 4; ++t) {
      const __bf16* vp = Vt + vtbase + (size_t)(t * 16 + lo) * Sc + j + hi * 16;
      v16bf vbf = cat16(*(const v8bf*)vp, *(const v8bf*)(vp + 8));
      acc[t] = wmma_bf16(pa, vbf, acc[t]);
    }
    __syncthreads();
  }

  // ---- epilogue: normalize by l (v_rcp), write bf16
  float rl[8];
#pragma unroll
  for (int r = 0; r < 8; ++r) rl[r] = fast_rcp(lr[r]);
#pragma unroll
  for (int t = 0; t < 4; ++t)
#pragma unroll
    for (int r = 0; r < 8; ++r) {
      const int row = q0 + r + hi * 8;
      O[base + (size_t)row * Dc + t * 16 + lo] = (__bf16)(acc[t][r] * rl[r]);
    }
}

// ---------------------------------------------------------------------------
extern "C" void kernel_launch(void* const* d_in, const int* in_sizes, int n_in,
                              void* d_out, int out_size, void* d_ws, size_t ws_size,
                              hipStream_t stream) {
  const float* x  = (const float*)d_in[0];
  const float* wq = (const float*)d_in[1];
  const float* wk = (const float*)d_in[2];
  const float* wv = (const float*)d_in[3];
  const float* wo = (const float*)d_in[4];

  const size_t nX = (size_t)Mrows * Dc;   // 4,194,304
  const size_t nW = (size_t)Dc * Dc;      // 1,048,576

  char* ws = (char*)d_ws;
  __bf16* xb  = (__bf16*)ws;  ws += nX * 2;
  __bf16* wqb = (__bf16*)ws;  ws += nW * 2;
  __bf16* wkb = (__bf16*)ws;  ws += nW * 2;
  __bf16* wvb = (__bf16*)ws;  ws += nW * 2;
  __bf16* wob = (__bf16*)ws;  ws += nW * 2;
  __bf16* qb  = (__bf16*)ws;  ws += nX * 2;
  __bf16* kb  = (__bf16*)ws;  ws += nX * 2;
  __bf16* vtb = (__bf16*)ws;  ws += nX * 2;   // V transposed per head
  __bf16* ab  = (__bf16*)ws;  ws += nX * 2;   // ~48 MB total

  // 1) cast inputs to bf16
  cast_f32_bf16<<<1024, 256, 0, stream>>>(x,  xb,  (int)nX);
  cast_f32_bf16<<<512,  256, 0, stream>>>(wq, wqb, (int)nW);
  cast_f32_bf16<<<512,  256, 0, stream>>>(wk, wkb, (int)nW);
  cast_f32_bf16<<<512,  256, 0, stream>>>(wv, wvb, (int)nW);
  cast_f32_bf16<<<512,  256, 0, stream>>>(wo, wob, (int)nW);

  // 2) q/k/v projections (q,k bf16 row-major; v written transposed per head)
  dim3 gg(Mrows / 16, Dc / 512);  // (256, 2)
  gemm_xwT<1><<<gg, 128, 0, stream>>>(xb, wqb, qb,  Mrows, Dc, Dc);
  gemm_xwT<1><<<gg, 128, 0, stream>>>(xb, wkb, kb,  Mrows, Dc, Dc);
  gemm_xwT<2><<<gg, 128, 0, stream>>>(xb, wvb, vtb, Mrows, Dc, Dc);

  // 3) QK RMSNorm per head (in place)
  const int nvec = Mrows * Hc;    // 65536
  rmsnorm_heads<<<nvec / 8, 256, 0, stream>>>(qb, nvec);
  rmsnorm_heads<<<nvec / 8, 256, 0, stream>>>(kb, nvec);

  // 4) causal softcapped flash attention
  dim3 ga(Sc / 16, Bc * Hc);      // (128, 32)
  flash_attn<<<ga, 32, 0, stream>>>(qb, kb, vtb, ab);

  // 5) output projection (f32 out)
  gemm_xwT<0><<<gg, 128, 0, stream>>>(ab, wob, d_out, Mrows, Dc, Dc);
}
